// BlockSparseAttention_57939108823582
// MI455X (gfx1250) — matrix operand order repaired and checked by reading, not compiled, not run on hardware
//
#include <hip/hip_runtime.h>

// Fused attention (B=8, S=4096, D=64, fp32) for MI455X / gfx1250.
// Compute-bound (~34 GFLOP, ~20MB working set fits in 192MB L2):
// QK^T and PV via v_wmma_f32_16x16x32_f16, flash-style online softmax in
// base-2 (scale folded into Q), fp32 accumulation.
//
// Key-order permutation trick: the K-dim of the PV WMMA is a reduction
// index, so A (V^T) and B (P^T) fragments use a common permuted key order
// chosen so P^T is a pure in-lane register repack (no cross-lane exchange)
// and V^T fragments stay contiguous 16B LDS reads.
//
// Workspace: Q/K/V in f16, 3 * 8*4096*64 * 2B = 12 MB of d_ws.

typedef _Float16 half_t;
typedef __attribute__((ext_vector_type(16))) _Float16 v16h;
typedef __attribute__((ext_vector_type(8)))  _Float16 v8h;
typedef __attribute__((ext_vector_type(8)))  float    v8f;

#define BATCH 8
#define SEQ   4096
#define DM    64
#define QTILE 128   // queries per workgroup = 8 waves * 16
#define KSTEP 64    // keys per inner step (two PV k-chunks of 32)

#define KT_STRIDE 80  // halves per K-tile row   (160B: 16B aligned, bank-spread)
#define VT_STRIDE 72  // halves per V^T-tile row (144B: 16B aligned, bank-spread)

union H16 { v16h v; v8h h[2]; };
union U32H { unsigned u; _Float16 h[2]; };

#define LOG2E 1.44269504088896340736f

// v_exp_f32 is natively 2^x on AMDGPU; use the direct builtin when available.
__device__ __forceinline__ float fast_exp2(float x) {
#if __has_builtin(__builtin_amdgcn_exp2f)
    return __builtin_amdgcn_exp2f(x);
#else
    return exp2f(x);
#endif
}

// ---------------------------------------------------------------------------
// Kernel 1: Q = (x*Wq^T + bq) * (1/8 * log2e), K, V -> f16 workspace.
// Block = 256 threads = 4 rows x 64 out-dims; x rows staged in LDS.
// ---------------------------------------------------------------------------
__global__ __launch_bounds__(256) void proj_qkv_kernel(
    const float* __restrict__ x,
    const float* __restrict__ Wq, const float* __restrict__ bq,
    const float* __restrict__ Wk, const float* __restrict__ bk,
    const float* __restrict__ Wv, const float* __restrict__ bv,
    half_t* __restrict__ Qh, half_t* __restrict__ Kh, half_t* __restrict__ Vh)
{
    __shared__ float xs[4 * 64];
    const int tid = threadIdx.x;
    xs[tid] = x[(size_t)blockIdx.x * 256 + tid];
    __syncthreads();

    const int r = tid >> 6;      // local row 0..3
    const int e = tid & 63;      // output dim
    const float* __restrict__ xr = &xs[r * 64];
    const float* __restrict__ wq = &Wq[e * 64];
    const float* __restrict__ wk = &Wk[e * 64];
    const float* __restrict__ wv = &Wv[e * 64];

    float aq = bq[e], ak = bk[e], av = bv[e];
#pragma unroll
    for (int i = 0; i < 64; ++i) {
        const float xi = xr[i];
        aq = __builtin_fmaf(xi, wq[i], aq);
        ak = __builtin_fmaf(xi, wk[i], ak);
        av = __builtin_fmaf(xi, wv[i], av);
    }
    const size_t row = (size_t)blockIdx.x * 4 + r;
    Qh[row * DM + e] = (half_t)(aq * (0.125f * LOG2E));  // 1/sqrt(64), base-2
    Kh[row * DM + e] = (half_t)ak;
    Vh[row * DM + e] = (half_t)av;
}

// ---------------------------------------------------------------------------
// Kernel 2: flash attention. grid = (SEQ/QTILE, BATCH), block = 256 (8 waves).
// Each wave owns a 16-query tile. S^T = K*Q^T so each lane holds all of its
// query's scores (softmax reduce = in-lane + one shfl_xor(16)).
// O^T = V^T * P^T with permuted key order (P^T fragment is register-only).
// ---------------------------------------------------------------------------
__global__ __launch_bounds__(256) void flash_attn_kernel(
    const half_t* __restrict__ Qh, const half_t* __restrict__ Kh,
    const half_t* __restrict__ Vh, float* __restrict__ out)
{
    __shared__ half_t Kt[KSTEP * KT_STRIDE];  // K tile, row-major [key][d]
    __shared__ half_t Vt[DM * VT_STRIDE];     // V^T tile        [d][key]

    const int tid  = threadIdx.x;
    const int lane = tid & 31;
    const int wave = tid >> 5;
    const int h    = lane >> 4;   // half-wave index
    const int n    = lane & 15;   // column index (query / d-local / key row)
    const int b    = blockIdx.y;
    const int q    = blockIdx.x * QTILE + wave * 16 + n;

    const size_t kvbase = (size_t)(b * SEQ) * DM;
    const half_t* __restrict__ Kg = Kh + kvbase;
    const half_t* __restrict__ Vg = Vh + kvbase;

    // --- Q as B-matrix fragments (held for whole K loop) -------------------
    // B layout: halves[0..7] = d {8h..8h+7}, halves[8..15] = {16+8h..}
    const half_t* __restrict__ Qrow = Qh + kvbase + (size_t)q * DM;
    H16 Bq0, Bq1;
    Bq0.h[0] = *(const v8h*)(Qrow + 8 * h);
    Bq0.h[1] = *(const v8h*)(Qrow + 16 + 8 * h);
    Bq1.h[0] = *(const v8h*)(Qrow + 32 + 8 * h);
    Bq1.h[1] = *(const v8h*)(Qrow + 48 + 8 * h);

    v8f acc[4] = {v8f{}, v8f{}, v8f{}, v8f{}};   // O^T, 4 d-tiles of 16
    float m_run = -__builtin_inff();
    float l_run = 0.0f;

    // staging: K -> 16 halves/thread (1 key, d-range of 16)
    //          V -> 2 keys x 8 d/thread, packed pairs into b32 LDS stores
    const int sk_key = tid >> 2;
    const int sk_d   = (tid & 3) * 16;
    const int sv_kp  = (tid >> 3) * 2;   // key pair base
    const int sv_d   = (tid & 7) * 8;

    v8h k0, k1, va, vb;
    {
        const half_t* kp = Kg + (size_t)sk_key * DM + sk_d;
        k0 = *(const v8h*)kp;  k1 = *(const v8h*)(kp + 8);
        const half_t* vp = Vg + (size_t)sv_kp * DM + sv_d;
        va = *(const v8h*)vp;  vb = *(const v8h*)(vp + DM);
    }

    for (int kb = 0; kb < SEQ; kb += KSTEP) {
        __syncthreads();  // previous iteration's LDS reads complete

        // --- stage current tile, then immediately issue next tile's loads --
        *(v8h*)(&Kt[sk_key * KT_STRIDE + sk_d])     = k0;
        *(v8h*)(&Kt[sk_key * KT_STRIDE + sk_d + 8]) = k1;
#pragma unroll
        for (int i = 0; i < 8; ++i) {
            U32H u; u.h[0] = va[i]; u.h[1] = vb[i];
            *(unsigned*)(&Vt[(sv_d + i) * VT_STRIDE + sv_kp]) = u.u;
        }
        if (kb + KSTEP < SEQ) {
            const half_t* kp = Kg + (size_t)(kb + KSTEP + sk_key) * DM + sk_d;
            k0 = *(const v8h*)kp;  k1 = *(const v8h*)(kp + 8);
            const half_t* vp = Vg + (size_t)(kb + KSTEP + sv_kp) * DM + sv_d;
            va = *(const v8h*)vp;  vb = *(const v8h*)(vp + DM);
        }
        __syncthreads();

        // --- S^T tiles: C[m=key][n=query], 4 key-tiles of 16 ---------------
        // A layout: halves[0..7] = d {8h..8h+7}, halves[8..15] = {16+8h..}
        v8f c[4];
#pragma unroll
        for (int t = 0; t < 4; ++t) {
            const half_t* Kr = &Kt[(t * 16 + n) * KT_STRIDE];
            H16 A0, A1;
            A0.h[0] = *(const v8h*)(Kr + 8 * h);
            A0.h[1] = *(const v8h*)(Kr + 16 + 8 * h);
            A1.h[0] = *(const v8h*)(Kr + 32 + 8 * h);
            A1.h[1] = *(const v8h*)(Kr + 48 + 8 * h);
            v8f z = {};
            z = __builtin_amdgcn_wmma_f32_16x16x32_f16(false, A0.v, false, Bq0.v,
                                                       (short)0, z, false, false);
            c[t] = __builtin_amdgcn_wmma_f32_16x16x32_f16(false, A1.v, false, Bq1.v,
                                                          (short)0, z, false, false);
        }
        // Lane holds, for query n: key 16t + 8h + j in c[t][j].

        // --- online softmax (base-2 domain) --------------------------------
        float mloc = c[0][0];
#pragma unroll
        for (int t = 0; t < 4; ++t)
#pragma unroll
            for (int j = 0; j < 8; ++j) mloc = fmaxf(mloc, c[t][j]);
        mloc = fmaxf(mloc, __shfl_xor(mloc, 16, 32));
        const float m_new = fmaxf(m_run, mloc);
        const float alpha = fast_exp2(m_run - m_new);  // 0 on first step

        float p[4][8], sloc = 0.0f;
#pragma unroll
        for (int t = 0; t < 4; ++t)
#pragma unroll
            for (int j = 0; j < 8; ++j) {
                p[t][j] = fast_exp2(c[t][j] - m_new);
                sloc += p[t][j];
            }
        sloc += __shfl_xor(sloc, 16, 32);
        l_run = l_run * alpha + sloc;
        m_run = m_new;

#pragma unroll
        for (int t = 0; t < 4; ++t)
#pragma unroll
            for (int j = 0; j < 8; ++j) acc[t][j] *= alpha;

        // --- P^T fragments: pure register repack (permuted key order) ------
        // kchunk0 keys {0..31}: halves[0..7]=c[0] keys (8h+j),
        //                       halves[8..15]=c[1] keys (16+8h+j)
        // kchunk1 keys {32..63}: from c[2], c[3].
        H16 Bp0, Bp1;
#pragma unroll
        for (int j = 0; j < 8; ++j) {
            Bp0.h[0][j] = (half_t)p[0][j];
            Bp0.h[1][j] = (half_t)p[1][j];
            Bp1.h[0][j] = (half_t)p[2][j];
            Bp1.h[1][j] = (half_t)p[3][j];
        }

        // --- O^T += V^T * P^T, 4 d-tiles x 2 k-chunks ----------------------
        // Same key order as P^T: halves[0..7] = keys 32c + 8h + 0..7,
        // halves[8..15] = keys 32c + 16 + 8h + 0..7.
#pragma unroll
        for (int t = 0; t < 4; ++t) {
            const half_t* Vr = &Vt[(t * 16 + n) * VT_STRIDE];
            H16 Av0, Av1;
            Av0.h[0] = *(const v8h*)(Vr + 8 * h);
            Av0.h[1] = *(const v8h*)(Vr + 16 + 8 * h);
            Av1.h[0] = *(const v8h*)(Vr + 32 + 8 * h);
            Av1.h[1] = *(const v8h*)(Vr + 48 + 8 * h);
            acc[t] = __builtin_amdgcn_wmma_f32_16x16x32_f16(
                false, Av0.v, false, Bp0.v, (short)0, acc[t], false, false);
            acc[t] = __builtin_amdgcn_wmma_f32_16x16x32_f16(
                false, Av1.v, false, Bp1.v, (short)0, acc[t], false, false);
        }
    }

    // --- finalize: out[b][q][d] = O^T[d][q] / l ----------------------------
    const float rl = 1.0f / l_run;
    float* __restrict__ orow = out + kvbase + (size_t)q * DM;
#pragma unroll
    for (int t = 0; t < 4; ++t) {
        v8f o;
#pragma unroll
        for (int j = 0; j < 8; ++j) o[j] = acc[t][j] * rl;
        *(v8f*)(orow + t * 16 + 8 * h) = o;  // d = t*16 + 8h + j, 32B aligned
    }
}

// ---------------------------------------------------------------------------
extern "C" void kernel_launch(void* const* d_in, const int* in_sizes, int n_in,
                              void* d_out, int out_size, void* d_ws, size_t ws_size,
                              hipStream_t stream) {
    (void)in_sizes; (void)n_in; (void)out_size; (void)ws_size;
    const float* x  = (const float*)d_in[0];
    const float* Wq = (const float*)d_in[1];
    const float* bq = (const float*)d_in[2];
    const float* Wk = (const float*)d_in[3];
    const float* bk = (const float*)d_in[4];
    const float* Wv = (const float*)d_in[5];
    const float* bv = (const float*)d_in[6];
    float* out = (float*)d_out;

    const size_t nelem = (size_t)BATCH * SEQ * DM;
    half_t* Qh = (half_t*)d_ws;          // 4 MB
    half_t* Kh = Qh + nelem;             // 4 MB
    half_t* Vh = Kh + nelem;             // 4 MB  (12 MB workspace total)

    proj_qkv_kernel<<<dim3(BATCH * SEQ / 4), dim3(256), 0, stream>>>(
        x, Wq, bq, Wk, bk, Wv, bv, Qh, Kh, Vh);
    flash_attn_kernel<<<dim3(SEQ / QTILE, BATCH), dim3(256), 0, stream>>>(
        Qh, Kh, Vh, out);
}
